// EdgeConv_71579924955361
// MI455X (gfx1250) — matrix operand order repaired and checked by reading, not compiled, not run on hardware
//
#include <hip/hip_runtime.h>
#include <hip/hip_fp16.h>

// ---------------------------------------------------------------------------
// EdgeConv, restructured:
//   Stage 1 (per node, N=100K):  U = x @ (W1a - W1b)^T + b1 ; V = x @ W1b^T
//   Stage 2 (per edge, E=1.6M):  out = relu( relu(U[row]+V[col]) @ W2^T + b2 )
// Stage 1: f32 WMMA 16x16x4 (f16 WMMA fallback if builtin absent).
// Stage 2: f16 WMMA 16x16x32, f32 accumulate. U/V (51 MB) stay in L2;
// output stream (410 MB) uses non-temporal stores.
// ---------------------------------------------------------------------------

typedef float    v2f  __attribute__((ext_vector_type(2)));
typedef float    v8f  __attribute__((ext_vector_type(8)));
typedef _Float16 v16h __attribute__((ext_vector_type(16)));

#ifndef __has_builtin
#define __has_builtin(x) 0
#endif

#if __has_builtin(__builtin_amdgcn_wmma_f32_16x16x4_f32)
#define HAVE_WMMA_F32X4 1
#else
#define HAVE_WMMA_F32X4 0
#endif

static __device__ __forceinline__ v8f wmma_f16(v16h a, v16h b, v8f c) {
  // D = A(16x32 f16) * B(32x16 f16) + C(16x16 f32)
  return __builtin_amdgcn_wmma_f32_16x16x32_f16(false, a, false, b, (short)0, c,
                                                false, false);
}

// ---------------------------------------------------------------------------
// Stage 1: one wave per 16-node tile. M = node-in-tile, N = out channel,
// K = 64 input channels.
// ---------------------------------------------------------------------------
__global__ void __launch_bounds__(256)
uv_precompute(const float* __restrict__ x, const float* __restrict__ W1,
              const float* __restrict__ b1, float* __restrict__ U,
              float* __restrict__ V, int nNodes, int nTiles) {
  const int lane  = threadIdx.x & 31;
  const int laneM = lane & 15;
  const int hi    = lane >> 4;
  const int wave  = (int)((blockIdx.x * blockDim.x + threadIdx.x) >> 5);
  if (wave >= nTiles) return;  // wave-uniform: EXEC stays all-ones for WMMA
  const int nb = wave * 16;
  int nrow = nb + laneM;
  if (nrow >= nNodes) nrow = nNodes - 1;  // clamp loads; stores are guarded

  const v8f z8 = {0.f, 0.f, 0.f, 0.f, 0.f, 0.f, 0.f, 0.f};
  v8f accU[4], accV[4];
#pragma unroll
  for (int nt = 0; nt < 4; ++nt) { accU[nt] = z8; accV[nt] = z8; }

#if HAVE_WMMA_F32X4
  // Full-precision path: 16 K-steps of 16x16x4 f32 WMMA per output N-tile.
  // A layout (16x4 f32): lanes 0-15 hold K=0,1 ; lanes 16-31 hold K=2,3.
#pragma unroll
  for (int ks = 0; ks < 16; ++ks) {
    const int k = ks * 4 + hi * 2;
    v2f a = *(const v2f*)(x + (size_t)nrow * 64 + k);
#pragma unroll
    for (int nt = 0; nt < 4; ++nt) {
      const int o = nt * 16 + laneM;  // B column = W1 row (out channel)
      v2f wa = *(const v2f*)(W1 + (size_t)o * 128 + k);        // W1a slice
      v2f wb = *(const v2f*)(W1 + (size_t)o * 128 + 64 + k);   // W1b slice
      v2f wd = wa - wb;
      accU[nt] = __builtin_amdgcn_wmma_f32_16x16x4_f32(
          false, a, false, wd, (short)0, accU[nt], false, false);
      accV[nt] = __builtin_amdgcn_wmma_f32_16x16x4_f32(
          false, a, false, wb, (short)0, accV[nt], false, false);
    }
  }
#else
  // f16 fallback (2 K-chunks of 32).
  v16h aF[2];
#pragma unroll
  for (int c = 0; c < 2; ++c) {
    const int base = c * 32 + hi * 8;
#pragma unroll
    for (int j = 0; j < 16; ++j) {
      const int k = (j < 8) ? (base + j) : (base + 16 + (j - 8));
      aF[c][j] = (_Float16)x[(size_t)nrow * 64 + k];
    }
  }
#pragma unroll
  for (int nt = 0; nt < 4; ++nt) {
    const int o = nt * 16 + laneM;
#pragma unroll
    for (int c = 0; c < 2; ++c) {
      v16h bU, bV;
#pragma unroll
      for (int j = 0; j < 16; ++j) {
        const int k = c * 32 + hi * 16 + j;
        float wa = W1[(size_t)o * 128 + k];
        float wb = W1[(size_t)o * 128 + 64 + k];
        bU[j] = (_Float16)(wa - wb);
        bV[j] = (_Float16)wb;
      }
      accU[nt] = wmma_f16(aF[c], bU, accU[nt]);
      accV[nt] = wmma_f16(aF[c], bV, accV[nt]);
    }
  }
#endif

  // D layout: VGPR r -> M = r + hi*8, N = laneM. Fold b1 into U.
#pragma unroll
  for (int nt = 0; nt < 4; ++nt) {
    const float bias = b1[nt * 16 + laneM];
#pragma unroll
    for (int r = 0; r < 8; ++r) {
      const int m = nb + hi * 8 + r;
      if (m < nNodes) {
        U[(size_t)m * 64 + nt * 16 + laneM] = accU[nt][r] + bias;
        V[(size_t)m * 64 + nt * 16 + laneM] = accV[nt][r];
      }
    }
  }
}

// ---------------------------------------------------------------------------
// Stage 2: persistent waves; one 16-edge tile per iteration.
// M = edge-in-tile, N = out channel (4 tiles of 16), K = 64 (2 chunks of 32).
// ---------------------------------------------------------------------------
__global__ void __launch_bounds__(256)
edge_mlp(const float* __restrict__ U, const float* __restrict__ V,
         const int* __restrict__ ei, const float* __restrict__ W2,
         const float* __restrict__ b2, float* __restrict__ out, int E,
         int nTiles) {
  const int lane   = threadIdx.x & 31;
  const int laneM  = lane & 15;
  const int hi     = lane >> 4;
  const int wave   = (int)((blockIdx.x * blockDim.x + threadIdx.x) >> 5);
  const int nWaves = (int)((gridDim.x * blockDim.x) >> 5);

  // Preload W2 fragments once per wave. B layout (32x16 f16): lane N = laneM,
  // lane's K set = {base..base+7} U {base+16..base+23}, base = hi*8 (matches A).
  v16h  wf[4][2];
  float b2v[4];
#pragma unroll
  for (int nt = 0; nt < 4; ++nt) {
    b2v[nt] = b2[nt * 16 + laneM];
#pragma unroll
    for (int c = 0; c < 2; ++c) {
      const float* wr = W2 + (size_t)(nt * 16 + laneM) * 64 + c * 32 + hi * 8;
#pragma unroll
      for (int j = 0; j < 16; ++j) wf[nt][c][j] = (_Float16)wr[(j < 8) ? j : (j + 8)];
    }
  }

  const v8f z8 = {0.f, 0.f, 0.f, 0.f, 0.f, 0.f, 0.f, 0.f};
  for (int t = wave; t < nTiles; t += nWaves) {
    const int eb = t * 16;
    int e = eb + laneM;
    if (e >= E) e = E - 1;  // clamp loads; stores guarded
    const int r  = __builtin_nontemporal_load(ei + e);             // row idx
    const int ci = __builtin_nontemporal_load(ei + (size_t)E + e); // col idx

    // Prefetch next tile's streaming index reads.
    const int tn = t + nWaves;
    if (tn < nTiles) {
      __builtin_prefetch(ei + (size_t)tn * 16 + laneM, 0, 0);
      __builtin_prefetch(ei + (size_t)E + (size_t)tn * 16 + laneM, 0, 0);
    }

    // Build A = relu(U[row] + V[col]) in f16, per ISA 16-bit A layout:
    // lane's K set = {base..base+7} U {base+16..base+23}, base = c*32 + hi*8.
    v16h aF[2];
#pragma unroll
    for (int ch = 0; ch < 2; ++ch) {
      const int    base = ch * 32 + hi * 8;
      const float* up   = U + (size_t)r * 64 + base;
      const float* vp   = V + (size_t)ci * 64 + base;
      float ub[16], vb[16];
      *(float4*)(&ub[0])  = *(const float4*)(up + 0);
      *(float4*)(&ub[4])  = *(const float4*)(up + 4);
      *(float4*)(&ub[8])  = *(const float4*)(up + 16);
      *(float4*)(&ub[12]) = *(const float4*)(up + 20);
      *(float4*)(&vb[0])  = *(const float4*)(vp + 0);
      *(float4*)(&vb[4])  = *(const float4*)(vp + 4);
      *(float4*)(&vb[8])  = *(const float4*)(vp + 16);
      *(float4*)(&vb[12]) = *(const float4*)(vp + 20);
#pragma unroll
      for (int j = 0; j < 16; ++j)
        aF[ch][j] = (_Float16)fmaxf(ub[j] + vb[j], 0.0f);
    }

#pragma unroll
    for (int nt = 0; nt < 4; ++nt) {
      v8f acc = z8;
      acc = wmma_f16(aF[0], wf[nt][0], acc);
      acc = wmma_f16(aF[1], wf[nt][1], acc);
      // D: VGPR rr -> M = rr + hi*8, N = laneM. Bias + relu + NT store.
#pragma unroll
      for (int rr = 0; rr < 8; ++rr) {
        const int m = eb + hi * 8 + rr;
        if (m < E) {
          float o = fmaxf(acc[rr] + b2v[nt], 0.0f);
          __builtin_nontemporal_store(o, out + (size_t)m * 64 + nt * 16 + laneM);
        }
      }
    }
  }
}

// ---------------------------------------------------------------------------
extern "C" void kernel_launch(void* const* d_in, const int* in_sizes, int n_in,
                              void* d_out, int out_size, void* d_ws,
                              size_t ws_size, hipStream_t stream) {
  const float* x  = (const float*)d_in[0];
  const int*   ei = (const int*)d_in[1];  // edge_index [2,E], int32
  const float* W1 = (const float*)d_in[2];  // [64,128]
  const float* b1 = (const float*)d_in[3];  // [64]
  const float* W2 = (const float*)d_in[4];  // [64,64]
  const float* b2 = (const float*)d_in[5];  // [64]
  float*       out = (float*)d_out;

  const int nNodes = in_sizes[0] / 64;
  const int E      = in_sizes[1] / 2;

  // Workspace: U,V as [nNodes,64] f32 each (51.2 MB for N=100K).
  float* U = (float*)d_ws;
  float* V = U + (size_t)nNodes * 64;

  const int tiles1  = (nNodes + 15) / 16;
  const int blocks1 = (tiles1 + 7) / 8;  // 8 waves (256 thr) per block
  uv_precompute<<<blocks1, 256, 0, stream>>>(x, W1, b1, U, V, nNodes, tiles1);

  const int tiles2  = (E + 15) / 16;
  int blocks2 = (tiles2 + 7) / 8;
  if (blocks2 > 1280) blocks2 = 1280;  // persistent waves, amortize W2 preload
  edge_mlp<<<blocks2, 256, 0, stream>>>(U, V, ei, W2, b2, out, E, tiles2);
}
